// LatentLayer_88441966559691
// MI455X (gfx1250) — matrix operand order repaired and checked by reading, not compiled, not run on hardware
//
#include <hip/hip_runtime.h>

// ---------------------------------------------------------------------------
// VQ codebook loss: dist = ||z||^2 + ||e||^2 - 2 z.e^T ; loss = mean_m min_n dist
// N=4096, M=4096, D=1024, fp32 inputs.
// GEMM core: bf16 hi/lo split ("fp32 emulation") on V_WMMA_F32_16X16X32_BF16:
//   z.e ~= zh.eh + zh.el + zl.eh   (error ~2^-16 relative; norms stay exact fp32)
// 3 bf16 K=32 WMMAs replace 8 fp32 K=4 WMMAs -> 2.67x fewer matrix ops at equal
// issue rate, much more at realistic bf16:fp32 matrix throughput ratios.
// e-tile (hi+lo, 256KB) in LDS shared by 4 waves; waves partition N; fused
// min epilogue (no 64MB dist matrix ever materialized).
// ---------------------------------------------------------------------------

typedef float  v8f   __attribute__((ext_vector_type(8)));
typedef __bf16 v8bf  __attribute__((ext_vector_type(8)));
typedef __bf16 v16bf __attribute__((ext_vector_type(16)));

#define N_PTS   4096
#define M_CODES 4096
#define ZDIM    1024

#define WAVES_PER_WG 4
#define COLS_PER_WG  64          // 4 col-tiles of 16
#define ROW_TILES    2           // 32 rows per wave-iteration
#define COL_TILES    4

static __device__ __forceinline__ unsigned short f2bf(float x) {
    unsigned int u = __float_as_uint(x);
    u += 0x7FFFu + ((u >> 16) & 1u);          // round-to-nearest-even
    return (unsigned short)(u >> 16);
}
static __device__ __forceinline__ float bf2f(unsigned short b) {
    return __uint_as_float(((unsigned int)b) << 16);
}
static __device__ __forceinline__ v16bf cat8(v8bf lo, v8bf hi) {
    return __builtin_shufflevector(lo, hi, 0, 1, 2, 3, 4, 5, 6, 7,
                                           8, 9, 10, 11, 12, 13, 14, 15);
}

// ---- Kernel P: split z into bf16 hi/lo planes (row-major) -----------------
__global__ void zsplit_kernel(const float* __restrict__ z,
                              unsigned short* __restrict__ zh,
                              unsigned short* __restrict__ zl) {
    int i = blockIdx.x * blockDim.x + threadIdx.x;     // N_PTS*ZDIM threads
    float x = z[i];
    unsigned short h = f2bf(x);
    zh[i] = h;
    zl[i] = f2bf(x - bf2f(h));
}

// ---- Kernel A: z_sq[n] = sum_k z[n][k]^2 (exact fp32) ---------------------
__global__ void zsq_kernel(const float* __restrict__ z, float* __restrict__ zsq) {
    int n = blockIdx.x * blockDim.x + threadIdx.x;
    if (n >= N_PTS) return;
    const float4* row = (const float4*)(z + (size_t)n * ZDIM);
    float s = 0.f;
    #pragma unroll 4
    for (int i = 0; i < ZDIM / 4; ++i) {
        float4 v = row[i];
        s += v.x * v.x + v.y * v.y + v.z * v.z + v.w * v.w;
    }
    zsq[n] = s;
}

// ---- Kernel B: per-code min over all N, bf16x3 WMMA GEMM + fused min ------
__launch_bounds__(WAVES_PER_WG * 32)
__global__ void vq_min_kernel(const unsigned short* __restrict__ zh,
                              const unsigned short* __restrict__ zl,
                              const float* __restrict__ e,
                              const float* __restrict__ zsq,
                              float* __restrict__ colmin) {
    extern __shared__ unsigned short smem_us[];
    unsigned short* ehs = smem_us;                              // [64][1024] bf16 hi
    unsigned short* els = smem_us + COLS_PER_WG * ZDIM;         // [64][1024] bf16 lo
    float* minbuf = (float*)(smem_us + 2 * COLS_PER_WG * ZDIM); // [4][64]

    const int tid   = threadIdx.x;
    const int wave  = tid >> 5;                                 // wave32
    const int lane  = tid & 31;
    const int mBase = blockIdx.x * COLS_PER_WG;

    // Stage + split the e-tile into LDS (hi and lo bf16 planes, row-major).
    for (int i = tid; i < COLS_PER_WG * ZDIM; i += WAVES_PER_WG * 32) {
        const int col = i >> 10;                                // i / ZDIM
        const int k   = i & (ZDIM - 1);
        float x = e[(size_t)(mBase + col) * ZDIM + k];
        unsigned short h = f2bf(x);
        ehs[col * ZDIM + k] = h;
        els[col * ZDIM + k] = f2bf(x - bf2f(h));
    }
    __syncthreads();

    // 16-bit fragment layouts used here (wave32; contiguous k, not ISA table order), lane = {mnIdx, phase p}:
    //   A(16x32): lane holds K = 16p+[0..15]                     (two 16B runs)
    //   B(32x16): lane holds K = 16p+[0..15]                     (one 32B run)
    //   C(16x16 f32): lane holds rows M = r + 8p, col = mnIdx
    const int mnIdx = lane & 15;
    const int ph    = lane >> 4;

    float runMin[COL_TILES] = {3.0e38f, 3.0e38f, 3.0e38f, 3.0e38f};

    // Waves partition N: wave w takes 32-row blocks w, w+4, w+8, ...
    const int nBlocks = N_PTS / (ROW_TILES * 16 * WAVES_PER_WG);   // 32 per wave
    for (int it = 0; it < nBlocks; ++it) {
        const int nBase = (it * WAVES_PER_WG + wave) * (ROW_TILES * 16);

        const unsigned short* zhR[ROW_TILES];
        const unsigned short* zlR[ROW_TILES];
        #pragma unroll
        for (int rt = 0; rt < ROW_TILES; ++rt) {
            const size_t roff = (size_t)(nBase + rt * 16 + mnIdx) * ZDIM;
            zhR[rt] = zh + roff;
            zlR[rt] = zl + roff;
        }

        v8f acc[ROW_TILES][COL_TILES] = {};        // 8 independent chains
        for (int k = 0; k < ZDIM; k += 32) {
            const int aOff = k + 16 * ph;
            const int bOff = k + 16 * ph;

            v16bf ah[ROW_TILES], al[ROW_TILES];
            #pragma unroll
            for (int rt = 0; rt < ROW_TILES; ++rt) {  // global_load_b128 x4 / rt
                ah[rt] = cat8(*(const v8bf*)(zhR[rt] + aOff),
                              *(const v8bf*)(zhR[rt] + aOff + 8));
                al[rt] = cat8(*(const v8bf*)(zlR[rt] + aOff),
                              *(const v8bf*)(zlR[rt] + aOff + 8));
            }
            v16bf bh[COL_TILES], bl[COL_TILES];
            #pragma unroll
            for (int ct = 0; ct < COL_TILES; ++ct) {  // 32B contiguous LDS loads
                const int crow = (ct * 16 + mnIdx) * ZDIM + bOff;
                bh[ct] = *(const v16bf*)(ehs + crow);
                bl[ct] = *(const v16bf*)(els + crow);
            }

            // Three emulation terms, grouped so each accumulator is revisited
            // only every 8 WMMAs (>= bf16 WMMA RAW hazard distance of 5).
            #pragma unroll
            for (int ct = 0; ct < COL_TILES; ++ct)
                #pragma unroll
                for (int rt = 0; rt < ROW_TILES; ++rt)
                    acc[rt][ct] = __builtin_amdgcn_wmma_f32_16x16x32_bf16(
                        false, ah[rt], false, bh[ct], (short)0, acc[rt][ct], false, false);
            #pragma unroll
            for (int ct = 0; ct < COL_TILES; ++ct)
                #pragma unroll
                for (int rt = 0; rt < ROW_TILES; ++rt)
                    acc[rt][ct] = __builtin_amdgcn_wmma_f32_16x16x32_bf16(
                        false, ah[rt], false, bl[ct], (short)0, acc[rt][ct], false, false);
            #pragma unroll
            for (int ct = 0; ct < COL_TILES; ++ct)
                #pragma unroll
                for (int rt = 0; rt < ROW_TILES; ++rt)
                    acc[rt][ct] = __builtin_amdgcn_wmma_f32_16x16x32_bf16(
                        false, al[rt], false, bh[ct], (short)0, acc[rt][ct], false, false);
        }

        // Epilogue: fold z_sq - 2*cross into per-column running min.
        const int nOff = nBase + ((lane < 16) ? 0 : 8);
        #pragma unroll
        for (int rt = 0; rt < ROW_TILES; ++rt) {
            #pragma unroll
            for (int r = 0; r < 8; ++r) {
                const float zq = zsq[nOff + rt * 16 + r];
                #pragma unroll
                for (int ct = 0; ct < COL_TILES; ++ct)
                    runMin[ct] = fminf(runMin[ct], zq - 2.0f * acc[rt][ct][r]);
            }
        }
    }

    // Lanes L and L+16 hold the same column: combine within wave.
    #pragma unroll
    for (int ct = 0; ct < COL_TILES; ++ct)
        runMin[ct] = fminf(runMin[ct], __shfl_xor(runMin[ct], 16, 32));

    // Each wave publishes its 64 partial column mins.
    if (lane < 16) {
        #pragma unroll
        for (int ct = 0; ct < COL_TILES; ++ct)
            minbuf[wave * COLS_PER_WG + ct * 16 + lane] = runMin[ct];
    }
    __syncthreads();

    // Reduce across the 4 waves, add exact fp32 ||e_m||^2, store.
    if (tid < COLS_PER_WG) {
        float m = fminf(fminf(minbuf[tid], minbuf[COLS_PER_WG + tid]),
                        fminf(minbuf[2 * COLS_PER_WG + tid], minbuf[3 * COLS_PER_WG + tid]));
        float esq = 0.f;
        const float4* erow = (const float4*)(e + (size_t)(mBase + tid) * ZDIM);
        #pragma unroll 4
        for (int i = 0; i < ZDIM / 4; ++i) {
            float4 v = erow[i];
            esq += v.x * v.x + v.y * v.y + v.z * v.z + v.w * v.w;
        }
        colmin[mBase + tid] = m + esq;
    }
}

// ---- Kernel C: mean of the 4096 per-code minima ---------------------------
__global__ void mean_kernel(const float* __restrict__ colmin, float* __restrict__ out) {
    __shared__ float red[256];
    float s = 0.f;
    for (int m = threadIdx.x; m < M_CODES; m += 256) s += colmin[m];
    red[threadIdx.x] = s;
    __syncthreads();
    for (int ofs = 128; ofs > 0; ofs >>= 1) {
        if (threadIdx.x < ofs) red[threadIdx.x] += red[threadIdx.x + ofs];
        __syncthreads();
    }
    if (threadIdx.x == 0) out[0] = red[0] / (float)M_CODES;
}

// ---------------------------------------------------------------------------
extern "C" void kernel_launch(void* const* d_in, const int* in_sizes, int n_in,
                              void* d_out, int out_size, void* d_ws, size_t ws_size,
                              hipStream_t stream) {
    const float* z = (const float*)d_in[0];   // [4096, 1024]
    const float* e = (const float*)d_in[1];   // [4096, 1024]

    // Workspace layout: zsq | colmin | zh | zl
    float* zsq    = (float*)d_ws;                              // 16 KB
    float* colmin = zsq + N_PTS;                               // 16 KB
    unsigned short* zh = (unsigned short*)(colmin + M_CODES);  // 8 MB
    unsigned short* zl = zh + (size_t)N_PTS * ZDIM;            // 8 MB

    zsplit_kernel<<<(N_PTS * ZDIM) / 256, 256, 0, stream>>>(z, zh, zl);
    zsq_kernel<<<N_PTS / 256, 256, 0, stream>>>(z, zsq);

    const size_t ldsBytes = (size_t)2 * COLS_PER_WG * ZDIM * sizeof(unsigned short) // 256 KB
                          + (size_t)WAVES_PER_WG * COLS_PER_WG * sizeof(float);     // 1 KB
    vq_min_kernel<<<M_CODES / COLS_PER_WG, WAVES_PER_WG * 32, ldsBytes, stream>>>(
        zh, zl, e, zsq, colmin);

    mean_kernel<<<1, 256, 0, stream>>>(colmin, (float*)d_out);
}
